// LFADS_Generator_SourceAttnCell_13391708029287
// MI455X (gfx1250) — matrix operand order repaired and checked by reading, not compiled, not run on hardware
//
#include <hip/hip_runtime.h>
#include <hip/hip_bf16.h>

// ---------------------------------------------------------------------------
// LFADS Generator + SourceAttn cell, fused f16-WMMA pipeline for gfx1250.
// All GEMMs run on v_wmma_f32_16x16x32_f16 (fp32 accumulate); elementwise
// math (sigmoid/tanh/clip/softmax) stays fp32. Attention uses a single-pass
// softmax over the S axis (scores are ReLU-bounded O(10), far below fp32 exp
// overflow; clamped at 60 for safety) with register-resident weight
// fragments. All GEMM tiles are full by construction, so EXEC is all-ones
// through every WMMA (ISA 7.12 requirement).
// ---------------------------------------------------------------------------

typedef __attribute__((ext_vector_type(16))) _Float16 v16h;
typedef __attribute__((ext_vector_type(8)))  _Float16 v8h;
typedef __attribute__((ext_vector_type(8)))  float    v8f;

static constexpr int Bsz   = 1024;  // batch
static constexpr int Ssz   = 64;    // #sources
static constexpr int INsz  = 256;   // input dim
static constexpr int Gsz   = 1024;  // generator dim
static constexpr int SRCsz = 256;   // source dim
static constexpr int Fsz   = 128;   // factors

// ---------------------------------------------------------------------------
// WMMA fragment loaders (this file's contiguous k = 16h+s order, same for A and B):
//  A 16x32 (MxK): lane m=lane&15; kb = (lane&16)?16:0; elems 0..7 -> K=kb..kb+7,
//                 elems 8..15 -> K=kb+8..kb+15  (two contiguous 16B chunks)
//  B 32x16 (KxN): lane n=lane&15; kb = (lane&16)?16:0; elems 0..15 -> K=kb..kb+15
//                 (one contiguous 32B chunk; W stored row-major (N,K))
// ---------------------------------------------------------------------------
__device__ __forceinline__ v16h load_a_frag(const _Float16* __restrict__ Atile,
                                            int ldk, int k0, int lane) {
  int m  = lane & 15;
  int kb = (lane & 16) ? 16 : 0;
  const _Float16* p = Atile + (size_t)m * ldk + k0 + kb;
  v8h lo = *reinterpret_cast<const v8h*>(p);
  v8h hi = *reinterpret_cast<const v8h*>(p + 8);
  v16h f;
#pragma unroll
  for (int i = 0; i < 8; ++i) { f[i] = lo[i]; f[i + 8] = hi[i]; }
  return f;
}

__device__ __forceinline__ v16h load_b_frag(const _Float16* __restrict__ Wtile,
                                            int ldk, int k0, int lane) {
  int n  = lane & 15;
  int kb = (lane & 16) ? 16 : 0;
  return *reinterpret_cast<const v16h*>(Wtile + (size_t)n * ldk + k0 + kb);
}

// ---------------------------------------------------------------------------
// Generic dual-source GEMM:  C(M,N) = A1(M,K1)@W1(N,K1)^T + A2@W2^T + bias
// Block: 256 threads = 8 waves, arranged (8/WN) x WN; each wave owns a 16x64
// strip (4 full C tiles, A-fragment reused 4x). No bounds guards: the grid
// tiles M and N exactly (all problem N are multiples of WN*64).
// Grid: (N/(WN*64), M/((8/WN)*16)).
// ---------------------------------------------------------------------------
template <int WN>
__global__ void gemm_dual_f16(const _Float16* __restrict__ A1,
                              const _Float16* __restrict__ W1, int K1,
                              const _Float16* __restrict__ A2,
                              const _Float16* __restrict__ W2, int K2,
                              const float* __restrict__ bias,
                              float* __restrict__ C, int N) {
  const int tid  = threadIdx.x;
  const int wave = tid >> 5;
  const int lane = tid & 31;
  const int row0 = blockIdx.y * ((8 / WN) * 16) + (wave / WN) * 16;
  const int col0 = blockIdx.x * (WN * 64) + (wave % WN) * 64;

  v8f acc[4] = {v8f{}, v8f{}, v8f{}, v8f{}};

  {  // pass 1
    const _Float16* At = A1 + (size_t)row0 * K1;
    for (int k0 = 0; k0 < K1; k0 += 32) {
      if (k0 + 64 < K1)
        __builtin_prefetch((const void*)(At + (size_t)(lane & 15) * K1 + k0 + 64), 0, 1);
      v16h aF = load_a_frag(At, K1, k0, lane);
#pragma unroll
      for (int t = 0; t < 4; ++t) {
        v16h bF = load_b_frag(W1 + (size_t)(col0 + t * 16) * K1, K1, k0, lane);
        acc[t] = __builtin_amdgcn_wmma_f32_16x16x32_f16(
            false, aF, false, bF, (short)0, acc[t], false, false);
      }
    }
  }
  if (A2) {  // pass 2 (accumulates into the same C tiles); uniform scalar branch
    const _Float16* At = A2 + (size_t)row0 * K2;
    for (int k0 = 0; k0 < K2; k0 += 32) {
      if (k0 + 64 < K2)
        __builtin_prefetch((const void*)(At + (size_t)(lane & 15) * K2 + k0 + 64), 0, 1);
      v16h aF = load_a_frag(At, K2, k0, lane);
#pragma unroll
      for (int t = 0; t < 4; ++t) {
        v16h bF = load_b_frag(W2 + (size_t)(col0 + t * 16) * K2, K2, k0, lane);
        acc[t] = __builtin_amdgcn_wmma_f32_16x16x32_f16(
            false, aF, false, bF, (short)0, acc[t], false, false);
      }
    }
  }

  // Epilogue. C/D layout: n = lane&15 within tile, m = i + 8*(lane>=16).
  const int mb = row0 + ((lane & 16) ? 8 : 0);
#pragma unroll
  for (int t = 0; t < 4; ++t) {
    int n = col0 + t * 16 + (lane & 15);
    float bv = bias ? bias[n] : 0.0f;
#pragma unroll
    for (int i = 0; i < 8; ++i)
      C[(size_t)(mb + i) * N + n] = acc[t][i] + bv;
  }
}

// ---------------------------------------------------------------------------
// Fused attention: per (16 b-rows x 16 e-cols) wave tile, loop s, score tile
// via WMMA against register-resident Wa_s fragments (hoisted; s-invariant),
// sc = relu(p + src@Wa_s^T), single-pass softmax over s (no running max:
// sc in [0, ~10] for this data; clamped at 60 << fp32 exp overflow),
// ctx = sum softmax * src.
// Block: 256 threads = 8 waves covering 128 e-cols. Grid: (B/16, SRC/128).
// Per-wave registers: 8 B-frags (64 VGPR) + state preg/den/num (24) + acc (8)
// + A-frag (8) -> fits without spills.
// ---------------------------------------------------------------------------
__global__ void attn_softmax_ctx(const _Float16* __restrict__ src_h,   // (S,B,SRC) f16
                                 const float*    __restrict__ src_f,   // (S,B,SRC) f32
                                 const float*    __restrict__ p,       // (B,SRC)   f32
                                 const _Float16* __restrict__ Was,     // (SRC,SRC) f16 (N,K)
                                 _Float16*       __restrict__ ctx_h) { // (B,SRC)   f16
  const int tid  = threadIdx.x;
  const int wave = tid >> 5;
  const int lane = tid & 31;
  const int b0   = blockIdx.x * 16;
  const int e0   = blockIdx.y * 128 + wave * 16;
  const int nlo  = lane & 15;
  const int mhi  = (lane & 16) ? 8 : 0;
  const int e    = e0 + nlo;

  // Hoist the wave's Wa_s fragments for all 8 k-steps (s-invariant).
  v16h bfr[8];
#pragma unroll
  for (int kk = 0; kk < 8; ++kk)
    bfr[kk] = load_b_frag(Was + (size_t)e0 * SRCsz, SRCsz, kk * 32, lane);

  float preg[8], den[8], num[8];
#pragma unroll
  for (int i = 0; i < 8; ++i) {
    preg[i] = p[(size_t)(b0 + mhi + i) * SRCsz + e];
    den[i]  = 0.0f;
    num[i]  = 0.0f;
  }

  for (int s = 0; s < Ssz; ++s) {
    const _Float16* At = src_h + ((size_t)s * Bsz + b0) * SRCsz;
    v8f acc = v8f{};
#pragma unroll
    for (int kk = 0; kk < 8; ++kk) {
      v16h aF = load_a_frag(At, SRCsz, kk * 32, lane);
      acc = __builtin_amdgcn_wmma_f32_16x16x32_f16(
          false, aF, false, bfr[kk], (short)0, acc, false, false);
    }
#pragma unroll
    for (int i = 0; i < 8; ++i) {
      float sc = fmaxf(acc[i] + preg[i], 0.0f);                        // relu score
      sc = fminf(sc, 60.0f);                                           // overflow guard
      float v  = src_f[((size_t)s * Bsz + b0 + mhi + i) * SRCsz + e];  // value (same elem)
      float w  = __expf(sc);
      den[i] += w;
      num[i] += w * v;
    }
  }

#pragma unroll
  for (int i = 0; i < 8; ++i)
    ctx_h[(size_t)(b0 + mhi + i) * SRCsz + e] = (_Float16)(num[i] / den[i]);
}

// ---------------------------------------------------------------------------
// Elementwise kernels (fp32 math)
// ---------------------------------------------------------------------------
__global__ void cvt_f32_f16(const float* __restrict__ s, int stride,
                            _Float16* __restrict__ d, long total, int cols) {
  for (long i = (long)blockIdx.x * blockDim.x + threadIdx.x; i < total;
       i += (long)gridDim.x * blockDim.x) {
    long r = i / cols;
    int  c = (int)(i - r * cols);
    d[i] = (_Float16)s[r * (long)stride + c];
  }
}

// r,u from ru; u_out fp32; rh = r*h as f16 (input to W_rhc GEMM)
__global__ void ew_ru(const float* __restrict__ ru, const float* __restrict__ h,
                      float* __restrict__ u_out, _Float16* __restrict__ rh_h) {
  const long total = (long)Bsz * Gsz;
  for (long i = (long)blockIdx.x * blockDim.x + threadIdx.x; i < total;
       i += (long)gridDim.x * blockDim.x) {
    long b = i / Gsz;
    int  g = (int)(i - b * Gsz);
    float rv = 1.0f / (1.0f + __expf(-ru[b * (2L * Gsz) + g]));
    float uv = 1.0f / (1.0f + __expf(-(ru[b * (2L * Gsz) + Gsz + g] + 1.0f)));
    u_out[i] = uv;
    rh_h[i]  = (_Float16)(rv * h[i]);
  }
}

// gen = clip(u*h + (1-u)*tanh(cpre), +-5); writes fp32 (d_out) + f16 copy
__global__ void ew_gen(const float* __restrict__ cpre, const float* __restrict__ u,
                       const float* __restrict__ h, float* __restrict__ gen_out,
                       _Float16* __restrict__ gen_h) {
  const long total = (long)Bsz * Gsz;
  for (long i = (long)blockIdx.x * blockDim.x + threadIdx.x; i < total;
       i += (long)gridDim.x * blockDim.x) {
    float c  = tanhf(cpre[i]);
    float uv = u[i];
    float g  = uv * h[i] + (1.0f - uv) * c;
    g = fminf(fmaxf(g, -5.0f), 5.0f);
    gen_out[i] = g;
    gen_h[i]   = (_Float16)g;
  }
}

__global__ void ew_tanh_h(const float* __restrict__ in, _Float16* __restrict__ out,
                          long total) {
  for (long i = (long)blockIdx.x * blockDim.x + threadIdx.x; i < total;
       i += (long)gridDim.x * blockDim.x)
    out[i] = (_Float16)tanhf(in[i]);
}

// ---------------------------------------------------------------------------
// Host orchestration
// ---------------------------------------------------------------------------
extern "C" void kernel_launch(void* const* d_in, const int* in_sizes, int n_in,
                              void* d_out, int out_size, void* d_ws, size_t ws_size,
                              hipStream_t stream) {
  const float* x       = (const float*)d_in[0];
  const float* h       = (const float*)d_in[1];
  const float* src     = (const float*)d_in[2];
  const float* W_xru   = (const float*)d_in[3];
  const float* W_hru   = (const float*)d_in[4];
  const float* b_hru   = (const float*)d_in[5];
  const float* W_xc    = (const float*)d_in[6];
  const float* W_rhc   = (const float*)d_in[7];
  const float* b_rhc   = (const float*)d_in[8];
  const float* W_alpha = (const float*)d_in[9];
  const float* W_out   = (const float*)d_in[10];
  const float* W_fac   = (const float*)d_in[11];

  char* wsp = (char*)d_ws;
  auto carve = [&](size_t bytes) -> void* {
    void* ptr = (void*)wsp;
    wsp += (bytes + 255) & ~(size_t)255;
    return ptr;
  };

  // f16 operand mirrors
  _Float16* x_h    = (_Float16*)carve((size_t)Bsz * INsz * 2);
  _Float16* h_h    = (_Float16*)carve((size_t)Bsz * Gsz * 2);
  _Float16* src_hh = (_Float16*)carve((size_t)Ssz * Bsz * SRCsz * 2);
  _Float16* Wxru_h = (_Float16*)carve((size_t)2 * Gsz * INsz * 2);
  _Float16* Whru_h = (_Float16*)carve((size_t)2 * Gsz * Gsz * 2);
  _Float16* Wxc_h  = (_Float16*)carve((size_t)Gsz * INsz * 2);
  _Float16* Wrhc_h = (_Float16*)carve((size_t)Gsz * Gsz * 2);
  _Float16* Wag_h  = (_Float16*)carve((size_t)SRCsz * Gsz * 2);
  _Float16* Was_h  = (_Float16*)carve((size_t)SRCsz * SRCsz * 2);
  _Float16* Wog_h  = (_Float16*)carve((size_t)Gsz * Gsz * 2);
  _Float16* Wos_h  = (_Float16*)carve((size_t)Gsz * SRCsz * 2);
  _Float16* Wfac_h = (_Float16*)carve((size_t)Fsz * Gsz * 2);
  _Float16* rh_h   = (_Float16*)carve((size_t)Bsz * Gsz * 2);
  _Float16* gen_h  = (_Float16*)carve((size_t)Bsz * Gsz * 2);
  _Float16* ctx_h  = (_Float16*)carve((size_t)Bsz * SRCsz * 2);
  _Float16* attn_h = (_Float16*)carve((size_t)Bsz * Gsz * 2);
  // f32 intermediates
  float* ru_f   = (float*)carve((size_t)Bsz * 2 * Gsz * 4);
  float* u_f    = (float*)carve((size_t)Bsz * Gsz * 4);
  float* cpre_f = (float*)carve((size_t)Bsz * Gsz * 4);
  float* p_f    = (float*)carve((size_t)Bsz * SRCsz * 4);
  float* apre_f = (float*)carve((size_t)Bsz * Gsz * 4);

  auto cvt = [&](const float* s, int stride, _Float16* d, long rows, int cols) {
    long total = rows * (long)cols;
    long gl = (total + 255) / 256;
    int  grid = (gl > 4096) ? 4096 : (int)gl;
    cvt_f32_f16<<<grid, 256, 0, stream>>>(s, stride, d, total, cols);
  };
  auto gemm = [&](const _Float16* A1, const _Float16* W1, int K1,
                  const _Float16* A2, const _Float16* W2, int K2,
                  const float* bias, float* C, int M, int N) {
    if (N % 256 == 0) {  // WN=4: block spans 256 cols x 32 rows
      dim3 g(N / 256, M / 32);
      gemm_dual_f16<4><<<g, 256, 0, stream>>>(A1, W1, K1, A2, W2, K2, bias, C, N);
    } else {             // WN=2: block spans 128 cols x 64 rows (N=128 case)
      dim3 g(N / 128, M / 64);
      gemm_dual_f16<2><<<g, 256, 0, stream>>>(A1, W1, K1, A2, W2, K2, bias, C, N);
    }
  };

  // --- stage 0: precision conversion (one pass; amortized over all GEMMs) ---
  cvt(x,             INsz,        x_h,    Bsz,            INsz);
  cvt(h,             Gsz,         h_h,    Bsz,            Gsz);
  cvt(src,           SRCsz,       src_hh, (long)Ssz * Bsz, SRCsz);
  cvt(W_xru,         INsz,        Wxru_h, 2L * Gsz,       INsz);
  cvt(W_hru,         Gsz,         Whru_h, 2L * Gsz,       Gsz);
  cvt(W_xc,          INsz,        Wxc_h,  Gsz,            INsz);
  cvt(W_rhc,         Gsz,         Wrhc_h, Gsz,            Gsz);
  cvt(W_alpha,       Gsz + SRCsz, Wag_h,  SRCsz,          Gsz);   // cols [0,G)
  cvt(W_alpha + Gsz, Gsz + SRCsz, Was_h,  SRCsz,          SRCsz); // cols [G,G+SRC)
  cvt(W_out,         Gsz + SRCsz, Wog_h,  Gsz,            Gsz);
  cvt(W_out + Gsz,   Gsz + SRCsz, Wos_h,  Gsz,            SRCsz);
  cvt(W_fac,         Gsz,         Wfac_h, Fsz,            Gsz);

  // --- stage 1: ru = x@Wxru^T + h@Whru^T + b_hru ---
  gemm(x_h, Wxru_h, INsz, h_h, Whru_h, Gsz, b_hru, ru_f, Bsz, 2 * Gsz);
  ew_ru<<<4096, 256, 0, stream>>>(ru_f, h, u_f, rh_h);

  // --- stage 2: cpre = x@Wxc^T + (r*h)@Wrhc^T + b_rhc ; gen ---
  gemm(x_h, Wxc_h, INsz, rh_h, Wrhc_h, Gsz, b_rhc, cpre_f, Bsz, Gsz);
  ew_gen<<<4096, 256, 0, stream>>>(cpre_f, u_f, h, (float*)d_out, gen_h);

  // --- stage 3: p = gen@Wa_g^T ; fused attention (scores->softmax->ctx) ---
  gemm(gen_h, Wag_h, Gsz, nullptr, nullptr, 0, nullptr, p_f, Bsz, SRCsz);
  attn_softmax_ctx<<<dim3(Bsz / 16, SRCsz / 128), 256, 0, stream>>>(
      src_hh, src, p_f, Was_h, ctx_h);

  // --- stage 4: attn_out = tanh(gen@Wo_g^T + ctx@Wo_s^T) ---
  gemm(gen_h, Wog_h, Gsz, ctx_h, Wos_h, SRCsz, nullptr, apre_f, Bsz, Gsz);
  ew_tanh_h<<<4096, 256, 0, stream>>>(apre_f, attn_h, (long)Bsz * Gsz);

  // --- stage 5: factors = attn_out@W_fac^T -> d_out[B*G:] ---
  gemm(attn_h, Wfac_h, Gsz, nullptr, nullptr, 0, nullptr,
       (float*)d_out + (size_t)Bsz * Gsz, Bsz, Fsz);
}
